// NeuroSAT_35519379538224
// MI455X (gfx1250) — compile-verified, matrix operand order checked by reading, not run on hardware
//
#include <hip/hip_runtime.h>
#include <cstddef>

// ---------------------------------------------------------------------------
// NeuroSAT on gfx1250 (MI455X). Dense math on v_wmma_f32_16x16x32_bf16
// (fp32 accumulate). ~160MB working set -> L2 resident -> compute bound, so:
//  * weights pre-packed once into per-fragment bf16 layout (coalesced 32B/lane)
//  * MLP weights staged whole in LDS (32KB), reused by all 8 waves of a block
//  * LSTM weights staged per-gate-step in LDS (32-48KB slices), 16x L2 reuse
//  * RT row-tiles per wave so each B fragment feeds RT WMMAs from registers
//  * MLP intermediates kept in bf16 (half the A traffic, no cvt on reload)
//  * LSTM gate math fused into the GEMM epilogue (no z buffer)
// ---------------------------------------------------------------------------

typedef __attribute__((ext_vector_type(16))) __bf16 v16bf;
typedef __attribute__((ext_vector_type(8)))  __bf16 v8bf;
typedef __attribute__((ext_vector_type(8)))  float  v8f;

#define FMDIM 128

__device__ __forceinline__ v8f wmma_bf16(v16bf a, v16bf b, v8f c) {
  // (neg_a, A, neg_b, B, c_mod, C, reuse_a, reuse_b)
  return __builtin_amdgcn_wmma_f32_16x16x32_bf16(false, a, false, b, (short)0, c,
                                                 false, false);
}

__device__ __forceinline__ float sigmoid_f(float x) {
  return 1.f / (1.f + __expf(-x));
}
__device__ __forceinline__ float tanh_f(float x) {
  float e = __expf(2.f * x);
  return 1.f - 2.f / (e + 1.f);  // robust for |x| large
}

// Row mapping for concat-along-K sources.
// 0: identity. 1: flip (literal negation swap). 2: m + nv.
__device__ __forceinline__ int map_row(int m, int mode, int nv) {
  if (mode == 1) return (m < nv) ? (m + nv) : (m - nv);
  if (mode == 2) return m + nv;
  return m;
}

// 16x32 bf16 A-fragment, rows m0..m0+15, k-step kk. CDNA5 16-bit A layout:
// lane (h=lane>>4, r=lane&15) -> row m0+r, K = 32kk+8h+j (j<8),
// K = 32kk+16+8h+(j-8) (j>=8): two contiguous 8-element chunks.
template <typename T>
__device__ __forceinline__ v16bf load_a_frag(const T* __restrict__ src,
                                             int mrow, int lda, int coff,
                                             int kk, int h) {
  const T* p = src + (size_t)mrow * lda + coff + 32 * kk + 8 * h;
  v16bf a;
  if constexpr (sizeof(T) == 2) {
    v8bf lo = *(const v8bf*)p;
    v8bf hi = *(const v8bf*)(p + 16);
#pragma unroll
    for (int j = 0; j < 8; ++j) { a[j] = lo[j]; a[8 + j] = hi[j]; }
  } else {
#pragma unroll
    for (int j = 0; j < 8; ++j) a[j] = (__bf16)p[j];
#pragma unroll
    for (int j = 0; j < 8; ++j) a[8 + j] = (__bf16)p[16 + j];
  }
  return a;
}

// ---------------------------------------------------------------------------
// Weight packer: fp32 [128*S, N] (S chunk pointers of 128 rows, stride N) ->
// bf16 fragment layout: frag[((t*4S + kk)*32 + lane)*16 + j], so a B-fragment
// load is one coalesced 32B read per lane. B (32x16) striping: lane (h, n)
// holds col 16t+n, K = 32kk + 16*(j>=8) + 8h + (j&7).
// ---------------------------------------------------------------------------
__global__ __launch_bounds__(256) void pack_wb(const float* __restrict__ s0,
                                               const float* __restrict__ s1,
                                               const float* __restrict__ s2,
                                               int S, int N,
                                               __bf16* __restrict__ out) {
  int tid = blockIdx.x * blockDim.x + threadIdx.x;
  int total = S * 128 * N;
  if (tid >= total) return;
  int j    = tid & 15;
  int lane = (tid >> 4) & 31;
  int frag = tid >> 9;
  int KK = 4 * S;
  int kk = frag % KK;
  int t  = frag / KK;
  int K   = 32 * kk + ((j >= 8) ? 16 : 0) + 8 * (lane >> 4) + (j & 7);
  int col = 16 * t + (lane & 15);
  const float* s = (K < 128) ? s0 : ((K < 256) ? s1 : s2);
  out[tid] = (__bf16)s[(size_t)(K & 127) * N + col];
}

// ---------------------------------------------------------------------------
// GEMM: Out[M,N] = act( sum_s A_s @ W_chunk_s + bias ). One wave owns RT
// 16-row tiles; STAGE=true stages the packed weights in LDS (shared by all 8
// waves), each fragment then feeds RT WMMAs from ds_load.
// ---------------------------------------------------------------------------
template <int S, int RT, bool RELU, bool STAGE, typename TIN, typename TOUT>
__global__ __launch_bounds__(256) void gemm_bias_act(
    const TIN* __restrict__ A0, const TIN* __restrict__ A1,
    const TIN* __restrict__ A2, int lda0, int lda1, int lda2, int coff0,
    int coff1, int coff2, int mode0, int mode1, int mode2, int nv,
    const __bf16* __restrict__ Bfrag, const float* __restrict__ bias,
    TOUT* __restrict__ Out, int M, int N) {
  extern __shared__ __bf16 smem[];
  if constexpr (STAGE) {
    int nvec = (S * 128 * N) >> 3;  // 16B vectors
    const v8bf* g = (const v8bf*)Bfrag;
    v8bf* d = (v8bf*)smem;
    for (int i = threadIdx.x; i < nvec; i += 256) d[i] = g[i];
    __syncthreads();
  }

  int lane = threadIdx.x & 31;
  int wv   = threadIdx.x >> 5;
  int tbase = (blockIdx.x * 8 + wv) * RT;
  int h = lane >> 4, r = lane & 15;

  const TIN* Aps[3]   = {A0, A1, A2};
  const int  ldas[3]  = {lda0, lda1, lda2};
  const int  coffs[3] = {coff0, coff1, coff2};
  const int  modes[3] = {mode0, mode1, mode2};

  bool  valid[RT];
  v16bf areg[RT][4 * S];
#pragma unroll
  for (int rt = 0; rt < RT; ++rt) {
    int m0 = (tbase + rt) * 16;
    valid[rt] = (m0 < M);
    int mm = valid[rt] ? (m0 + r) : 0;  // clamp: safe read, store suppressed
#pragma unroll
    for (int s = 0; s < S; ++s) {
      int mrow = map_row(mm, modes[s], nv);
#pragma unroll
      for (int kk = 0; kk < 4; ++kk)
        areg[rt][s * 4 + kk] =
            load_a_frag(Aps[s], mrow, ldas[s], coffs[s], kk, h);
    }
  }

  const int NT = N >> 4;
  const int KK = 4 * S;
  for (int t = 0; t < NT; ++t) {
    v8f acc[RT];
#pragma unroll
    for (int rt = 0; rt < RT; ++rt)
      acc[rt] = (v8f){0.f, 0.f, 0.f, 0.f, 0.f, 0.f, 0.f, 0.f};
#pragma unroll
    for (int kk = 0; kk < KK; ++kk) {
      size_t fi = (size_t)(t * KK + kk) * 32 + lane;
      v16bf b;
      if constexpr (STAGE) b = ((const v16bf*)smem)[fi];
      else                 b = ((const v16bf*)Bfrag)[fi];
#pragma unroll
      for (int rt = 0; rt < RT; ++rt)
        acc[rt] = wmma_bf16(areg[rt][kk], b, acc[rt]);
    }
    float bv = bias[16 * t + r];
#pragma unroll
    for (int rt = 0; rt < RT; ++rt) {
      if (!valid[rt]) continue;
      int m0 = (tbase + rt) * 16;
#pragma unroll
      for (int v = 0; v < 8; ++v) {
        float x = acc[rt][v] + bv;
        if (RELU) x = fmaxf(x, 0.f);
        Out[(size_t)(m0 + v + 8 * h) * N + 16 * t + r] = (TOUT)x;
      }
    }
  }
}

// ---------------------------------------------------------------------------
// Fused LSTM: z[M,512] = sum_s A_s[M,128] @ W_chunk_s + b; gate tiles
// (t, t+8, t+16, t+24) accumulated together -> c/h updated in registers.
// Per-t staging: the 4*KK fragments needed for gate-step t (32-48KB) are
// copied to LDS by the whole block, giving 8 waves x RT tiles = 16x reuse of
// every L2 fetch. Two barriers per t; all waves participate (no early exit).
// ---------------------------------------------------------------------------
template <int S, int RT>
__global__ __launch_bounds__(256) void lstm_fused(
    const float* __restrict__ A0, const float* __restrict__ A1,
    const float* __restrict__ A2, int mode1, int nv,
    const __bf16* __restrict__ Bfrag, const float* __restrict__ bias,
    const float* __restrict__ Cin, float* __restrict__ Cout,
    float* __restrict__ Hout, int M) {
  extern __shared__ __bf16 sB[];  // 4*KK*512 bf16 per gate-step
  int lane = threadIdx.x & 31;
  int wv   = threadIdx.x >> 5;
  int tbase = (blockIdx.x * 8 + wv) * RT;
  int h = lane >> 4, r = lane & 15;

  const float* Aps[3]  = {A0, A1, A2};
  const int    modes[3] = {0, mode1, 0};
  const int    KK = 4 * S;

  bool  valid[RT];
  v16bf areg[RT][4 * S];
#pragma unroll
  for (int rt = 0; rt < RT; ++rt) {
    int m0 = (tbase + rt) * 16;
    valid[rt] = (m0 < M);
    int mm = valid[rt] ? (m0 + r) : 0;  // clamp: safe read, store suppressed
#pragma unroll
    for (int s = 0; s < S; ++s) {
      int mrow = map_row(mm, modes[s], nv);
#pragma unroll
      for (int kk = 0; kk < 4; ++kk)
        areg[rt][s * 4 + kk] = load_a_frag(Aps[s], mrow, FMDIM, 0, kk, h);
    }
  }

  const v8bf* Bv = (const v8bf*)Bfrag;
  for (int t = 0; t < 8; ++t) {
    // ---- stage the 4*KK fragments for gate-step t (gates i,f,g,o are
    //      global n-tiles t, t+8, t+16, t+24) ----
    __syncthreads();  // previous iteration done reading sB
    {
      v8bf* d = (v8bf*)sB;
      int nvec = 4 * KK * 64;  // frags * (512 bf16 / 8)
      for (int i = threadIdx.x; i < nvec; i += 256) {
        int g  = i / (KK * 64);
        int rm = i - g * (KK * 64);
        int kk = rm >> 6;
        int vv = rm & 63;
        d[i] = Bv[(size_t)((g * 8 + t) * KK + kk) * 64 + vv];
      }
    }
    __syncthreads();

    v8f ai[RT], af[RT], ag[RT], ao[RT];
#pragma unroll
    for (int rt = 0; rt < RT; ++rt) {
      ai[rt] = (v8f){0.f, 0.f, 0.f, 0.f, 0.f, 0.f, 0.f, 0.f};
      af[rt] = ai[rt]; ag[rt] = ai[rt]; ao[rt] = ai[rt];
    }
    const v16bf* Bs = (const v16bf*)sB;
#pragma unroll
    for (int kk = 0; kk < KK; ++kk) {
      v16bf bi = Bs[(size_t)(0 * KK + kk) * 32 + lane];
      v16bf bf = Bs[(size_t)(1 * KK + kk) * 32 + lane];
      v16bf bg = Bs[(size_t)(2 * KK + kk) * 32 + lane];
      v16bf bo = Bs[(size_t)(3 * KK + kk) * 32 + lane];
#pragma unroll
      for (int rt = 0; rt < RT; ++rt) {
        v16bf a = areg[rt][kk];
        ai[rt] = wmma_bf16(a, bi, ai[rt]);
        af[rt] = wmma_bf16(a, bf, af[rt]);
        ag[rt] = wmma_bf16(a, bg, ag[rt]);
        ao[rt] = wmma_bf16(a, bo, ao[rt]);
      }
    }
    float bi_s = bias[t * 16 + r];
    float bf_s = bias[128 + t * 16 + r];
    float bg_s = bias[256 + t * 16 + r];
    float bo_s = bias[384 + t * 16 + r];
#pragma unroll
    for (int rt = 0; rt < RT; ++rt) {
      if (!valid[rt]) continue;
      int m0 = (tbase + rt) * 16;
#pragma unroll
      for (int v = 0; v < 8; ++v) {
        size_t idx = (size_t)(m0 + v + 8 * h) * FMDIM + 16 * t + r;
        float c_old = Cin[idx];
        float iv = sigmoid_f(ai[rt][v] + bi_s);
        float fv = sigmoid_f(af[rt][v] + bf_s);
        float gv = tanh_f(ag[rt][v] + bg_s);
        float ov = sigmoid_f(ao[rt][v] + bo_s);
        float cn = fv * c_old + iv * gv;
        Cout[idx] = cn;
        Hout[idx] = ov * tanh_f(cn);
      }
    }
  }
}

// ---------------------------------------------------------------------------
// Segment scatter-add: out[dst[e], :] += pre[src[e], :] (bf16 in, fp32 accum).
// ---------------------------------------------------------------------------
__global__ __launch_bounds__(256) void seg_scatter(
    const __bf16* __restrict__ pre, const int* __restrict__ src_idx,
    const int* __restrict__ dst_idx, float* __restrict__ out, int n_edges) {
  long long gid = (long long)blockIdx.x * blockDim.x + threadIdx.x;
  int e = (int)(gid >> 7), col = (int)(gid & 127);
  if (e >= n_edges) return;
  float v = (float)pre[(size_t)src_idx[e] * FMDIM + col];
  __hip_atomic_fetch_add(&out[(size_t)dst_idx[e] * FMDIM + col], v,
                         __ATOMIC_RELAXED, __HIP_MEMORY_SCOPE_AGENT);
}

__global__ __launch_bounds__(256) void zero_buf(float* __restrict__ p,
                                                long long n) {
  long long i = (long long)blockIdx.x * blockDim.x + threadIdx.x;
  if (i < n) p[i] = 0.f;
}

// H[m,:] = init[:] / sqrt(128); C[m,:] = 0
__global__ __launch_bounds__(256) void init_state(
    const float* __restrict__ initvec, float* __restrict__ H,
    float* __restrict__ C, int M) {
  long long gid = (long long)blockIdx.x * blockDim.x + threadIdx.x;
  if (gid >= (long long)M * FMDIM) return;
  int f = (int)(gid & 127);
  H[gid] = initvec[f] * 0.08838834764831845f;  // 1/sqrt(128)
  C[gid] = 0.f;
}

// Final vote layer: [M,256](bf16) @ [256,1] + b -> [M] fp32
__global__ __launch_bounds__(256) void vote_out(const __bf16* __restrict__ X,
                                                const float* __restrict__ w,
                                                const float* __restrict__ b,
                                                float* __restrict__ out, int M) {
  int m = blockIdx.x * blockDim.x + threadIdx.x;
  if (m >= M) return;
  float acc = b[0];
#pragma unroll 8
  for (int k = 0; k < 256; ++k) acc += (float)X[(size_t)m * 256 + k] * w[k];
  out[m] = acc;
}

// ---------------------------------------------------------------------------
extern "C" void kernel_launch(void* const* d_in, const int* in_sizes, int n_in,
                              void* d_out, int out_size, void* d_ws,
                              size_t ws_size, hipStream_t stream) {
  const int NV = 10000, NL = 20000, NC = 42000;
  const int NE = in_sizes[26];  // 126000
  const int ROUNDS = 16;

  // Input order: setup_inputs() dict insertion order, depth-first.
  const float* L_init = (const float*)d_in[0];
  const float* C_init = (const float*)d_in[1];
  const float* lc_w[3] = {(const float*)d_in[2], (const float*)d_in[3],
                          (const float*)d_in[4]};
  const float* lc_b[3] = {(const float*)d_in[5], (const float*)d_in[6],
                          (const float*)d_in[7]};
  const float* cl_w[3] = {(const float*)d_in[8], (const float*)d_in[9],
                          (const float*)d_in[10]};
  const float* cl_b[3] = {(const float*)d_in[11], (const float*)d_in[12],
                          (const float*)d_in[13]};
  const float* Lk = (const float*)d_in[14];  // (256,512)
  const float* Lr = (const float*)d_in[15];  // (128,512)
  const float* Lb = (const float*)d_in[16];  // (512,)
  const float* Ck = (const float*)d_in[17];  // (128,512)
  const float* Cr = (const float*)d_in[18];  // (128,512)
  const float* Cb = (const float*)d_in[19];  // (512,)
  const float* vw[3] = {(const float*)d_in[20], (const float*)d_in[21],
                        (const float*)d_in[22]};
  const float* vb[3] = {(const float*)d_in[23], (const float*)d_in[24],
                        (const float*)d_in[25]};
  const int* edge_lit = (const int*)d_in[26];
  const int* edge_clause = (const int*)d_in[27];
  float* out = (float*)d_out;

  // -- workspace partition (fp32 blocks, then bf16 fragment blocks) --
  float* w = (float*)d_ws;
  size_t o = 0;
  auto A = [&](size_t n) { float* p = w + o; o += n; return p; };
  float* l_h[2] = {A((size_t)NL * 128), A((size_t)NL * 128)};
  float* l_c    = A((size_t)NL * 128);
  float* c_h[2] = {A((size_t)NC * 128), A((size_t)NC * 128)};
  float* c_c    = A((size_t)NC * 128);
  float* tmpA   = A((size_t)NC * 128);  // used as bf16 [*,128|256]
  float* tmpB   = A((size_t)NC * 128);  // used as bf16
  float* msgs   = A((size_t)NC * 128);  // fp32 accumulators
  __bf16* tA16 = (__bf16*)tmpA;
  __bf16* tB16 = (__bf16*)tmpB;
  __bf16* fr = (__bf16*)(w + o);
  size_t fo = 0;
  auto F = [&](size_t n) { __bf16* p = fr + fo; fo += n; return p; };
  __bf16* fr_lc[3] = {F(128 * 128), F(128 * 128), F(128 * 128)};
  __bf16* fr_cl[3] = {F(128 * 128), F(128 * 128), F(128 * 128)};
  __bf16* fr_CL = F(256 * 512);   // [Ck; Cr]
  __bf16* fr_LL = F(384 * 512);   // [Lk[:128]; Lk[128:]; Lr]
  __bf16* fr_v0 = F(256 * 256);
  __bf16* fr_v1 = F(256 * 256);

  auto pack = [&](const float* s0, const float* s1, const float* s2, int S,
                  int N, __bf16* dst) {
    int total = S * 128 * N;
    pack_wb<<<(total + 255) / 256, 256, 0, stream>>>(s0, s1, s2, S, N, dst);
  };
  for (int i = 0; i < 3; ++i) pack(lc_w[i], nullptr, nullptr, 1, 128, fr_lc[i]);
  for (int i = 0; i < 3; ++i) pack(cl_w[i], nullptr, nullptr, 1, 128, fr_cl[i]);
  pack(Ck, Cr, nullptr, 2, 512, fr_CL);
  pack(Lk, Lk + 128 * 512, Lr, 3, 512, fr_LL);
  pack(vw[0], vw[0] + 128 * 256, nullptr, 2, 256, fr_v0);
  pack(vw[1], vw[1] + 128 * 256, nullptr, 2, 256, fr_v1);

  init_state<<<((long long)NL * 128 + 255) / 256, 256, 0, stream>>>(
      L_init, l_h[0], l_c, NL);
  init_state<<<((long long)NC * 128 + 255) / 256, 256, 0, stream>>>(
      C_init, c_h[0], c_c, NC);

  // grids: 8 waves/block, RT row-tiles (16 rows) per wave
  auto grid_rt = [](int M, int RT) {
    int tiles = M / 16;
    int per_block = 8 * RT;
    return (tiles + per_block - 1) / per_block;
  };
  const int MLP_LDS  = 128 * 128 * 2;      // 32KB whole-weight staging
  const int LSTM2_LDS = 4 * 8 * 512 * 2;   // 32KB per gate-step (S=2)
  const int LSTM3_LDS = 4 * 12 * 512 * 2;  // 48KB per gate-step (S=3)
  int lh = 0, ch = 0;

  for (int rnd = 0; rnd < ROUNDS; ++rnd) {
    // literal MLP (128->128->128, relu/relu/linear), bf16 intermediates
    gemm_bias_act<1, 4, true, true, float, __bf16>
        <<<grid_rt(NL, 4), 256, MLP_LDS, stream>>>(
            l_h[lh], nullptr, nullptr, 128, 0, 0, 0, 0, 0, 0, 0, 0, NV,
            fr_lc[0], lc_b[0], tA16, NL, 128);
    gemm_bias_act<1, 4, true, true, __bf16, __bf16>
        <<<grid_rt(NL, 4), 256, MLP_LDS, stream>>>(
            tA16, nullptr, nullptr, 128, 0, 0, 0, 0, 0, 0, 0, 0, NV, fr_lc[1],
            lc_b[1], tB16, NL, 128);
    gemm_bias_act<1, 4, false, true, __bf16, __bf16>
        <<<grid_rt(NL, 4), 256, MLP_LDS, stream>>>(
            tB16, nullptr, nullptr, 128, 0, 0, 0, 0, 0, 0, 0, 0, NV, fr_lc[2],
            lc_b[2], tA16, NL, 128);
    // segment_sum lits -> clauses
    zero_buf<<<((long long)NC * 128 + 255) / 256, 256, 0, stream>>>(
        msgs, (long long)NC * 128);
    seg_scatter<<<(int)(((long long)NE * 128 + 255) / 256), 256, 0, stream>>>(
        tA16, edge_lit, edge_clause, msgs, NE);
    // clause LSTM: z = msgs@Ck + c_h@Cr + Cb
    lstm_fused<2, 2><<<grid_rt(NC, 2), 256, LSTM2_LDS, stream>>>(
        msgs, c_h[ch], nullptr, 0, NV, fr_CL, Cb, c_c, c_c, c_h[ch ^ 1], NC);
    ch ^= 1;
    // clause MLP
    gemm_bias_act<1, 4, true, true, float, __bf16>
        <<<grid_rt(NC, 4), 256, MLP_LDS, stream>>>(
            c_h[ch], nullptr, nullptr, 128, 0, 0, 0, 0, 0, 0, 0, 0, NV,
            fr_cl[0], cl_b[0], tA16, NC, 128);
    gemm_bias_act<1, 4, true, true, __bf16, __bf16>
        <<<grid_rt(NC, 4), 256, MLP_LDS, stream>>>(
            tA16, nullptr, nullptr, 128, 0, 0, 0, 0, 0, 0, 0, 0, NV, fr_cl[1],
            cl_b[1], tB16, NC, 128);
    gemm_bias_act<1, 4, false, true, __bf16, __bf16>
        <<<grid_rt(NC, 4), 256, MLP_LDS, stream>>>(
            tB16, nullptr, nullptr, 128, 0, 0, 0, 0, 0, 0, 0, 0, NV, fr_cl[2],
            cl_b[2], tA16, NC, 128);
    // segment_sum clauses -> lits
    zero_buf<<<((long long)NL * 128 + 255) / 256, 256, 0, stream>>>(
        msgs, (long long)NL * 128);
    seg_scatter<<<(int)(((long long)NE * 128 + 255) / 256), 256, 0, stream>>>(
        tA16, edge_clause, edge_lit, msgs, NE);
    // literal LSTM: z = msgs@Lk[:128] + flip(l_h)@Lk[128:] + l_h@Lr + Lb
    lstm_fused<3, 2><<<grid_rt(NL, 2), 256, LSTM3_LDS, stream>>>(
        msgs, l_h[lh], l_h[lh], /*mode1=flip*/ 1, NV, fr_LL, Lb, l_c, l_c,
        l_h[lh ^ 1], NL);
    lh ^= 1;
  }

  // vote MLP: variables = [l_h[:NV], l_h[NV:]] (concat along K via sources)
  gemm_bias_act<2, 2, true, false, float, __bf16>
      <<<grid_rt(NV, 2), 256, 0, stream>>>(
          l_h[lh], l_h[lh], nullptr, 128, 128, 0, 0, 0, 0, /*modes*/ 0, 2, 0,
          NV, fr_v0, vb[0], tA16, NV, 256);
  gemm_bias_act<2, 2, true, false, __bf16, __bf16>
      <<<grid_rt(NV, 2), 256, 0, stream>>>(
          tA16, tA16, nullptr, 256, 256, 0, /*coff*/ 0, 128, 0, 0, 0, 0, NV,
          fr_v1, vb[1], tB16, NV, 256);
  vote_out<<<(NV + 255) / 256, 256, 0, stream>>>(tB16, vw[2], vb[2], out, NV);
}
